// GATModule_88811333746750
// MI455X (gfx1250) — compile-verified, matrix operand order checked by reading, not run on hardware
//
#include <hip/hip_runtime.h>

#define NN  50000      // nodes
#define EE  800000     // raw edges
#define ETOT 850000    // edges + self loops
#define D1  64
#define H1  4
#define HD1 256        // H1*D1
#define D2  64
#define NEG 0.2f

typedef __attribute__((ext_vector_type(2)))  float    v2f;
typedef __attribute__((ext_vector_type(8)))  float    v8f;
typedef __attribute__((ext_vector_type(16))) _Float16 v16h;

#if __has_builtin(__builtin_amdgcn_wmma_f32_16x16x4_f32)
#define USE_WMMA_F32 1
#else
#define USE_WMMA_F32 0
#endif

// ---------------- fills ----------------
__global__ __launch_bounds__(256) void fill_f32(float* p, int n, float v) {
  int i = blockIdx.x * 256 + threadIdx.x;
  if (i < n) p[i] = v;
}
__global__ __launch_bounds__(256) void fill_u32(unsigned* p, int n, unsigned v) {
  int i = blockIdx.x * 256 + threadIdx.x;
  if (i < n) p[i] = v;
}

// ---------------- WMMA GEMM: C[M,Nc] = A[M,K] @ B[K,Nc], row-major fp32 ----
// One wave per 16x16 output tile. M must be a multiple of 16 (50000 = 3125*16),
// Nc a multiple of 16, K a multiple of 4 (f32 path) / 32 (f16 fallback).
__global__ __launch_bounds__(256) void gemm_wmma(const float* __restrict__ A,
                                                 const float* __restrict__ B,
                                                 float* __restrict__ C,
                                                 int M, int K, int Nc) {
  int wave = (blockIdx.x * blockDim.x + threadIdx.x) >> 5;
  int lane = threadIdx.x & 31;
  int tn = Nc >> 4;
  int tm = M >> 4;
  if (wave >= tm * tn) return;          // uniform per wave -> EXEC all 1s inside
  int mt = wave / tn, nt = wave % tn;
  int half = lane >> 4;                 // lane group 0..15 vs 16..31
  int lm   = lane & 15;

  v8f c = {};
  const float* arow = A + (size_t)(mt * 16 + lm) * K;
  const float* bcol = B + nt * 16 + lm;

#if USE_WMMA_F32
  // A 16x4 layout: lanes0-15 v0/v1 = K0/K1, lanes16-31 v0/v1 = K2/K3 (ISA 7.12.2)
  for (int k0 = 0; k0 < K; k0 += 4) {
    int ka = k0 + half * 2;
    v2f a, b;
    a.x = arow[ka];
    a.y = arow[ka + 1];
    b.x = bcol[(size_t)ka * Nc];
    b.y = bcol[(size_t)(ka + 1) * Nc];
    c = __builtin_amdgcn_wmma_f32_16x16x4_f32(false, a, false, b, (short)0, c,
                                              false, false);
  }
#else
  // f16 fallback: 16-bit A 16x32 layout (ISA 7.12.2), B mirrored.
  for (int k0 = 0; k0 < K; k0 += 32) {
    v16h a, b;
#pragma unroll
    for (int p = 0; p < 8; ++p) {
      int k = k0 + (p & 3) * 2 + half * 8 + (p >> 2) * 16;
      a[2 * p]     = (_Float16)arow[k];
      a[2 * p + 1] = (_Float16)arow[k + 1];
      b[2 * p]     = (_Float16)bcol[(size_t)k * Nc];
      b[2 * p + 1] = (_Float16)bcol[(size_t)(k + 1) * Nc];
    }
    c = __builtin_amdgcn_wmma_f32_16x16x32_f16(false, a, false, b, (short)0, c,
                                               false, false);
  }
#endif
  // C/D layout: VGPR r -> row M=r (lanes 0-15) / M=r+8 (lanes 16-31), N=lane%16
#pragma unroll
  for (int r = 0; r < 8; ++r) {
    int row = mt * 16 + r + half * 8;
    C[(size_t)row * Nc + nt * 16 + lm] = c[r];
  }
}

// --------- attention logits: wave per (node, head), D=64 --------------------
__global__ __launch_bounds__(256) void attn_logits(const float* __restrict__ h,
                                                   const float* __restrict__ avs,
                                                   const float* __restrict__ avd,
                                                   float* __restrict__ als,
                                                   float* __restrict__ ald,
                                                   int N, int H) {
  int wave = (blockIdx.x * 256 + threadIdx.x) >> 5;
  int lane = threadIdx.x & 31;
  if (wave >= N * H) return;
  int hh = wave % H;
  const float* row = h + (size_t)wave * 64;  // wave = n*H+hh, contiguous [N,H,64]
  float s = row[lane] * avs[hh * 64 + lane] +
            row[lane + 32] * avs[hh * 64 + lane + 32];
  float d = row[lane] * avd[hh * 64 + lane] +
            row[lane + 32] * avd[hh * 64 + lane + 32];
  for (int off = 16; off > 0; off >>= 1) {
    s += __shfl_down(s, off);
    d += __shfl_down(d, off);
  }
  if (lane == 0) { als[wave] = s; ald[wave] = d; }
}

// monotonic uint encoding so unsigned atomicMax == float max
__device__ __forceinline__ unsigned fkey(float f) {
  unsigned u = __float_as_uint(f);
  return (u & 0x80000000u) ? ~u : (u | 0x80000000u);
}
__device__ __forceinline__ float funkey(unsigned u) {
  return __uint_as_float((u & 0x80000000u) ? (u & 0x7fffffffu) : ~u);
}
__device__ __forceinline__ void edge_sd(const long long* ei, int e, int& s, int& d) {
  if (e < EE) { s = (int)ei[e]; d = (int)ei[EE + e]; }
  else        { s = d = e - EE; }   // appended self loops
}

__global__ __launch_bounds__(256) void edge_max(const long long* __restrict__ ei,
                                                const float* __restrict__ als,
                                                const float* __restrict__ ald,
                                                unsigned* mkey, int H) {
  int e = blockIdx.x * 256 + threadIdx.x;
  if (e >= ETOT) return;
  int s, d; edge_sd(ei, e, s, d);
  for (int h = 0; h < H; ++h) {
    float t = als[s * H + h] + ald[d * H + h];
    t = t > 0.f ? t : NEG * t;
    atomicMax(&mkey[d * H + h], fkey(t));
  }
}

__global__ __launch_bounds__(256) void edge_expsum(const long long* __restrict__ ei,
                                                   const float* __restrict__ als,
                                                   const float* __restrict__ ald,
                                                   const unsigned* __restrict__ mkey,
                                                   float* __restrict__ ex,
                                                   float* den, int H) {
  int e = blockIdx.x * 256 + threadIdx.x;
  if (e >= ETOT) return;
  int s, d; edge_sd(ei, e, s, d);
  for (int h = 0; h < H; ++h) {
    float t = als[s * H + h] + ald[d * H + h];
    t = t > 0.f ? t : NEG * t;
    float m = funkey(mkey[d * H + h]);
    float x = expf(t - m);
    ex[(size_t)e * H + h] = x;
    atomicAdd(&den[d * H + h], x);
  }
}

// scatter: 256 threads = 4 edges x 64 feature lanes (coalesced over features)
__global__ __launch_bounds__(256) void edge_scatter(const long long* __restrict__ ei,
                                                    const float* __restrict__ hsrc,
                                                    const float* __restrict__ ex,
                                                    const float* __restrict__ den,
                                                    float* acc, int H) {
  int e   = blockIdx.x * 4 + (threadIdx.x >> 6);
  int dth = threadIdx.x & 63;
  if (e >= ETOT) return;
  int s, d; edge_sd(ei, e, s, d);
  for (int h = 0; h < H; ++h) {
    float alpha = ex[(size_t)e * H + h] / (den[d * H + h] + 1e-16f);
    float v = hsrc[(size_t)s * H * 64 + h * 64 + dth] * alpha;
    atomicAdd(&acc[(size_t)d * H * 64 + h * 64 + dth], v);
  }
}

__global__ __launch_bounds__(256) void bias_act(float* a, const float* __restrict__ b,
                                                int n, int cols, int do_relu) {
  int i = blockIdx.x * 256 + threadIdx.x;
  if (i >= n) return;
  float v = a[i] + b[i % cols];
  a[i] = do_relu ? fmaxf(v, 0.f) : v;
}

// ---------------------------------------------------------------------------
extern "C" void kernel_launch(void* const* d_in, const int* in_sizes, int n_in,
                              void* d_out, int out_size, void* d_ws, size_t ws_size,
                              hipStream_t stream) {
  (void)in_sizes; (void)n_in; (void)out_size; (void)ws_size;
  const float*     x   = (const float*)d_in[0];
  const long long* ei  = (const long long*)d_in[1];   // int64 [2, EE]
  const float*     W1  = (const float*)d_in[2];
  const float*     as1 = (const float*)d_in[3];
  const float*     ad1 = (const float*)d_in[4];
  const float*     b1  = (const float*)d_in[5];
  const float*     W2  = (const float*)d_in[6];
  const float*     as2 = (const float*)d_in[7];
  const float*     ad2 = (const float*)d_in[8];
  const float*     b2  = (const float*)d_in[9];
  float*           out = (float*)d_out;

  // workspace layout (layer-2 buffers overlay h1, which is dead after scatter1)
  char*  base = (char*)d_ws;
  size_t off  = 0;
  auto carve = [&](size_t bytes) -> char* {
    char* r = base + off;
    off = (off + bytes + 255) & ~(size_t)255;
    return r;
  };
  float*    h1   = (float*)carve((size_t)NN * HD1 * 4);   // 51.2 MB
  float*    acc1 = (float*)carve((size_t)NN * HD1 * 4);   // 51.2 MB
  float*    alS1 = (float*)carve((size_t)NN * H1 * 4);
  float*    alD1 = (float*)carve((size_t)NN * H1 * 4);
  unsigned* m1   = (unsigned*)carve((size_t)NN * H1 * 4);
  float*    den1 = (float*)carve((size_t)NN * H1 * 4);
  float*    ex1  = (float*)carve((size_t)ETOT * H1 * 4);  // 13.6 MB
  // layer-2 temporaries reuse h1's region (17 MB needed < 51.2 MB available)
  char*  b2base = (char*)h1;
  size_t off2   = 0;
  auto carve2 = [&](size_t bytes) -> char* {
    char* r = b2base + off2;
    off2 = (off2 + bytes + 255) & ~(size_t)255;
    return r;
  };
  float*    h2   = (float*)carve2((size_t)NN * D2 * 4);
  float*    alS2 = (float*)carve2((size_t)NN * 4);
  float*    alD2 = (float*)carve2((size_t)NN * 4);
  unsigned* m2   = (unsigned*)carve2((size_t)NN * 4);
  float*    den2 = (float*)carve2((size_t)NN * 4);
  float*    ex2  = (float*)carve2((size_t)ETOT * 4);

  auto cdiv = [](int a, int b) { return (a + b - 1) / b; };
  int eblk = cdiv(ETOT, 256);
  int sblk = cdiv(ETOT, 4);

  // ---- layer 1 ----
  fill_f32<<<cdiv(NN * HD1, 256), 256, 0, stream>>>(acc1, NN * HD1, 0.f);
  fill_u32<<<cdiv(NN * H1, 256), 256, 0, stream>>>(m1, NN * H1, 0u);
  fill_f32<<<cdiv(NN * H1, 256), 256, 0, stream>>>(den1, NN * H1, 0.f);

  {
    int waves = (NN / 16) * (HD1 / 16);
    gemm_wmma<<<cdiv(waves * 32, 256), 256, 0, stream>>>(x, W1, h1, NN, D1, HD1);
  }
  attn_logits<<<cdiv(NN * H1 * 32, 256), 256, 0, stream>>>(h1, as1, ad1, alS1, alD1, NN, H1);
  edge_max   <<<eblk, 256, 0, stream>>>(ei, alS1, alD1, m1, H1);
  edge_expsum<<<eblk, 256, 0, stream>>>(ei, alS1, alD1, m1, ex1, den1, H1);
  edge_scatter<<<sblk, 256, 0, stream>>>(ei, h1, ex1, den1, acc1, H1);
  bias_act<<<cdiv(NN * HD1, 256), 256, 0, stream>>>(acc1, b1, NN * HD1, HD1, 1);

  // ---- layer 2 ----
  fill_u32<<<cdiv(NN, 256), 256, 0, stream>>>(m2, NN, 0u);
  fill_f32<<<cdiv(NN, 256), 256, 0, stream>>>(den2, NN, 0.f);
  fill_f32<<<cdiv(NN * D2, 256), 256, 0, stream>>>(out, NN * D2, 0.f);

  {
    int waves = (NN / 16) * (D2 / 16);
    gemm_wmma<<<cdiv(waves * 32, 256), 256, 0, stream>>>(acc1, W2, h2, NN, HD1, D2);
  }
  attn_logits<<<cdiv(NN * 32, 256), 256, 0, stream>>>(h2, as2, ad2, alS2, alD2, NN, 1);
  edge_max   <<<eblk, 256, 0, stream>>>(ei, alS2, alD2, m2, 1);
  edge_expsum<<<eblk, 256, 0, stream>>>(ei, alS2, alD2, m2, ex2, den2, 1);
  edge_scatter<<<sblk, 256, 0, stream>>>(ei, h2, ex2, den2, out, 1);
  bias_act<<<cdiv(NN * D2, 256), 256, 0, stream>>>(out, b2, NN * D2, D2, 0);
}
